// MoELayer_55611236548626
// MI455X (gfx1250) — matrix operand order repaired and checked by reading, not compiled, not run on hardware
//
#include <hip/hip_runtime.h>
#include <math.h>

// ---------------------------------------------------------------------------
// MoE layer for MI455X (gfx1250, wave32, WMMA).
// HBM roofline: 537 MB fp32 expert weights @ 23.3 TB/s (~23 us) dominates;
// compute (1.1 TFLOP) is far under the bf16 WMMA roof, so spend the headroom
// on a bf16x3 split (near-fp32 accuracy) on v_wmma_f32_16x16x32_bf16.
// 256-thread WGs (8 wave32s) in a 4(M)x2(N) wave grid; 128x128 tiles,
// K-step 32; fp32->bf16(hi/lo) converted ONCE per tile into LDS and
// double-buffered. Each wave owns a 32x64 sub-tile (2 A-frags x 4 B-frags),
// minimizing LDS fragment re-reads (24 ds_load_b128 : 24 WMMA per K-step).
// ---------------------------------------------------------------------------

#define D_MODEL 2048
#define D_HID   4096
#define NEXP    8
#define TOKENS  4096   // B*S = 2*2048

typedef __attribute__((ext_vector_type(16))) __bf16 v16bf;
typedef __attribute__((ext_vector_type(8)))  __bf16 v8bf;   // 16 bytes
typedef __attribute__((ext_vector_type(8)))  float  v8f;

union Frag { v16bf full; v8bf half[2]; };

// LDS layout per buffer (bytes): AHI[8192] ALO[8192] BHI[8192] BLO[8192]
#define LBUF   32768
#define AHI_OF 0
#define ALO_OF 8192
#define BHI_OF 16384
#define BLO_OF 24576

// ---- packed bf16 helpers ---------------------------------------------------
// hi_pack(x,y): [bf16(y) : bf16(x)] via one v_perm_b32 (truncation split; the
// lo term recovers the dropped bits, total error ~2^-16 relative).
__device__ __forceinline__ unsigned hi_pack(float x, float y) {
  return __builtin_amdgcn_perm(__float_as_uint(y), __float_as_uint(x), 0x07060302u);
}
__device__ __forceinline__ float lo_of(float x) {
  return x - __uint_as_float(__float_as_uint(x) & 0xffff0000u);  // exact
}

__device__ __forceinline__ float gelu_exact(float v) {
  return 0.5f * v * (1.0f + erff(v * 0.70710678118654752f));
}

// ---- Router: logits -> softmax -> top-2 -> renormalized combine[T,E] ------
__global__ __launch_bounds__(256)
void router_kernel(const float* __restrict__ x, const float* __restrict__ rw,
                   float* __restrict__ combine) {
  int t = blockIdx.x * blockDim.x + threadIdx.x;
  if (t >= TOKENS) return;
  const float* xr = x + (size_t)t * D_MODEL;
  float l[NEXP];
#pragma unroll
  for (int e = 0; e < NEXP; ++e) l[e] = 0.0f;
  for (int d = 0; d < D_MODEL; ++d) {
    float xv = xr[d];
    const float* rwd = rw + (size_t)d * NEXP;
#pragma unroll
    for (int e = 0; e < NEXP; ++e) l[e] = fmaf(xv, rwd[e], l[e]);
  }
  float m = l[0];
#pragma unroll
  for (int e = 1; e < NEXP; ++e) m = fmaxf(m, l[e]);
  float p[NEXP], s = 0.0f;
#pragma unroll
  for (int e = 0; e < NEXP; ++e) { p[e] = expf(l[e] - m); s += p[e]; }
  float inv = 1.0f / s;
#pragma unroll
  for (int e = 0; e < NEXP; ++e) p[e] *= inv;
  int i0 = 0;
#pragma unroll
  for (int e = 1; e < NEXP; ++e) if (p[e] > p[i0]) i0 = e;
  int i1 = (i0 == 0) ? 1 : 0;
#pragma unroll
  for (int e = 0; e < NEXP; ++e) if (e != i1 && e != i0 && p[e] > p[i1]) i1 = e;
  float wsum = p[i0] + p[i1];
  float* c = combine + (size_t)t * NEXP;
#pragma unroll
  for (int e = 0; e < NEXP; ++e) c[e] = 0.0f;
  c[i0] = p[i0] / wsum;
  c[i1] = p[i1] / wsum;
}

__global__ __launch_bounds__(256)
void zero_kernel(float* __restrict__ p, int n) {
  int i = blockIdx.x * blockDim.x + threadIdx.x;
  if (i < n) p[i] = 0.0f;
}

// ---- staging helpers -------------------------------------------------------
// A tile: 128 rows x 32 k, fp32 row-major in global -> bf16 hi/lo row-major in
// LDS (64 B per row). 1024 float4s / 256 threads = 4 each.
__device__ __forceinline__ void stage_a_load(const float* __restrict__ A, int lda,
                                             int mt, int k0, int tid, float4 r[4]) {
#pragma unroll
  for (int i = 0; i < 4; ++i) {
    int c = tid + 256 * i;
    int row = c >> 3, kq = (c & 7) * 4;
    r[i] = *(const float4*)(A + (size_t)(mt + row) * (size_t)lda + k0 + kq);
  }
}
__device__ __forceinline__ void stage_a_store(unsigned char* __restrict__ lds,
                                              int tid, const float4 r[4]) {
#pragma unroll
  for (int i = 0; i < 4; ++i) {
    int c = tid + 256 * i;
    int row = c >> 3, kq = (c & 7) * 4;
    int off = row * 64 + kq * 2;
    uint2 hi = make_uint2(hi_pack(r[i].x, r[i].y), hi_pack(r[i].z, r[i].w));
    uint2 lo = make_uint2(hi_pack(lo_of(r[i].x), lo_of(r[i].y)),
                          hi_pack(lo_of(r[i].z), lo_of(r[i].w)));
    *(uint2*)(lds + AHI_OF + off) = hi;
    *(uint2*)(lds + ALO_OF + off) = lo;
  }
}

// B tile: 32 k x 128 cols fp32 row-major in global -> bf16 hi/lo TRANSPOSED in
// LDS ([col][k], 64 B per col). Each thread: 4x4 micro-tile (4 k-rows x 4 cols).
__device__ __forceinline__ void stage_b_load(const float* __restrict__ B, int ldb,
                                             int nt, int k0, int tid, float4 r[4]) {
  int colg = (tid & 31) * 4;
  int kg   = (tid >> 5) * 4;
#pragma unroll
  for (int i = 0; i < 4; ++i)
    r[i] = *(const float4*)(B + (size_t)(k0 + kg + i) * (size_t)ldb + nt + colg);
}
__device__ __forceinline__ void stage_b_store(unsigned char* __restrict__ lds,
                                              int tid, const float4 r[4]) {
  int colg = (tid & 31) * 4;
  int kg   = (tid >> 5) * 4;
  float v[4][4];
#pragma unroll
  for (int i = 0; i < 4; ++i) {
    v[i][0] = r[i].x; v[i][1] = r[i].y; v[i][2] = r[i].z; v[i][3] = r[i].w;
  }
#pragma unroll
  for (int c = 0; c < 4; ++c) {
    int off = (colg + c) * 64 + kg * 2;
    uint2 hi = make_uint2(hi_pack(v[0][c], v[1][c]), hi_pack(v[2][c], v[3][c]));
    uint2 lo = make_uint2(hi_pack(lo_of(v[0][c]), lo_of(v[1][c])),
                          hi_pack(lo_of(v[2][c]), lo_of(v[3][c])));
    *(uint2*)(lds + BHI_OF + off) = hi;
    *(uint2*)(lds + BLO_OF + off) = lo;
  }
}

// ---- fragment loads from LDS (two ds_load_b128 per fragment half) ---------
__device__ __forceinline__ void load_a_frag(const unsigned char* lds, int rowbase,
                                            int lane, Frag& hi, Frag& lo) {
  int row = rowbase + (lane & 15);
  int sel = (lane < 16) ? 0 : 16;      // K {0-7,16-23} vs {8-15,24-31}
  int base = row * 64 + sel;
  hi.half[0] = *(const v8bf*)(lds + AHI_OF + base);
  hi.half[1] = *(const v8bf*)(lds + AHI_OF + base + 32);
  lo.half[0] = *(const v8bf*)(lds + ALO_OF + base);
  lo.half[1] = *(const v8bf*)(lds + ALO_OF + base + 32);
}
__device__ __forceinline__ void load_b_frag(const unsigned char* lds, int colbase,
                                            int lane, Frag& hi, Frag& lo) {
  int col = colbase + (lane & 15);
  int sel = (lane < 16) ? 0 : 16;      // K {0-7,16-23} vs {8-15,24-31}
  int base = col * 64 + sel;
  hi.half[0] = *(const v8bf*)(lds + BHI_OF + base);
  hi.half[1] = *(const v8bf*)(lds + BHI_OF + base + 32);
  lo.half[0] = *(const v8bf*)(lds + BLO_OF + base);
  lo.half[1] = *(const v8bf*)(lds + BLO_OF + base + 32);
}

// ---- Fused expert GEMM ----------------------------------------------------
// MODE 0: C = gelu(A @ B + bias)                 (writes H scratch)
// MODE 1: out += combine[:,e] * (A @ B + bias)   (accumulates output)
// 256 threads = 8 wave32s in a 4(M)x2(N) grid; tile 128x128; each wave owns
// 32(M)x64(N): acc[2][4], bf16x3 = Ahi*Bhi + Ahi*Blo + Alo*Bhi.
template <int MODE>
__global__ __launch_bounds__(256)
void moe_gemm_kernel(const float* __restrict__ A, const float* __restrict__ B,
                     const float* __restrict__ bias,
                     const float* __restrict__ combine, int expert,
                     float* __restrict__ C, int M, int N, int K) {
  __shared__ __align__(16) unsigned char smem[2 * LBUF];
  const int tid  = threadIdx.x;
  const int lane = tid & 31;
  const int wave = tid >> 5;
  const int mw   = wave & 3;   // M wave group: rows  mw*32 .. +32
  const int nw   = wave >> 2;  // N wave group: cols  nw*64 .. +64
  const int mt = blockIdx.x * 128;
  const int nt = blockIdx.y * 128;

  v8f acc[2][4];
#pragma unroll
  for (int i = 0; i < 2; ++i)
#pragma unroll
    for (int j = 0; j < 4; ++j) acc[i][j] = v8f{};

  const int nk = K / 32;
  float4 a_reg[4], b_reg[4];

  // prologue: stage k-step 0 into buffer 0
  stage_a_load(A, K, mt, 0, tid, a_reg);
  stage_b_load(B, N, nt, 0, tid, b_reg);
  stage_a_store(smem, tid, a_reg);
  stage_b_store(smem, tid, b_reg);
  __syncthreads();

#pragma unroll 1
  for (int kk = 0; kk < nk; ++kk) {
    unsigned char* cur = smem + (kk & 1) * LBUF;
    if (kk + 1 < nk) {
      int k1 = (kk + 1) * 32;
      stage_a_load(A, K, mt, k1, tid, a_reg);
      stage_b_load(B, N, nt, k1, tid, b_reg);
      if (kk + 2 < nk)   // prefetch following B slab (global_prefetch_b8)
        __builtin_prefetch(B + (size_t)((kk + 2) * 32 + (tid >> 5)) * (size_t)N
                             + nt + (tid & 31) * 4, 0, 1);
    }

    Frag ahi[2], alo[2];
#pragma unroll
    for (int mf = 0; mf < 2; ++mf)
      load_a_frag(cur, mw * 32 + mf * 16, lane, ahi[mf], alo[mf]);
#pragma unroll
    for (int nf = 0; nf < 4; ++nf) {
      Frag bhi, blo;
      load_b_frag(cur, nw * 64 + nf * 16, lane, bhi, blo);
#pragma unroll
      for (int mf = 0; mf < 2; ++mf) {
        acc[mf][nf] = __builtin_amdgcn_wmma_f32_16x16x32_bf16(
            false, ahi[mf].full, false, bhi.full, (short)0, acc[mf][nf], false, false);
        acc[mf][nf] = __builtin_amdgcn_wmma_f32_16x16x32_bf16(
            false, ahi[mf].full, false, blo.full, (short)0, acc[mf][nf], false, false);
        acc[mf][nf] = __builtin_amdgcn_wmma_f32_16x16x32_bf16(
            false, alo[mf].full, false, bhi.full, (short)0, acc[mf][nf], false, false);
      }
    }

    if (kk + 1 < nk) {
      unsigned char* nxt = smem + ((kk + 1) & 1) * LBUF;
      stage_a_store(nxt, tid, a_reg);
      stage_b_store(nxt, tid, b_reg);
    }
    __syncthreads();
  }

  // Epilogue per CDNA5 C layout: lanes 0-15 -> N=lane, M=r; lanes 16-31 -> M=r+8
  const int colb   = nt + nw * 64 + (lane & 15);
  const int rowoff = (lane < 16) ? 0 : 8;
#pragma unroll
  for (int mf = 0; mf < 2; ++mf) {
    const int m0 = mt + mw * 32 + mf * 16;
#pragma unroll
    for (int nf = 0; nf < 4; ++nf) {
      int col = colb + nf * 16;
      float bv = bias[col];
#pragma unroll
      for (int r = 0; r < 8; ++r) {
        int row = m0 + rowoff + r;
        float v = acc[mf][nf][r] + bv;
        size_t idx = (size_t)row * (size_t)N + col;
        if (MODE == 0) {
          C[idx] = gelu_exact(v);
        } else {
          float cw = combine[(size_t)row * NEXP + expert];
          C[idx] = C[idx] + cw * v;   // stream-ordered per-expert accumulation
        }
      }
    }
  }
}

// ---------------------------------------------------------------------------
extern "C" void kernel_launch(void* const* d_in, const int* in_sizes, int n_in,
                              void* d_out, int out_size, void* d_ws, size_t ws_size,
                              hipStream_t stream) {
  const float* x  = (const float*)d_in[0];  // [4096, 2048]
  const float* rw = (const float*)d_in[1];  // [2048, 8]
  const float* w1 = (const float*)d_in[2];  // [8, 2048, 4096]
  const float* b1 = (const float*)d_in[3];  // [8, 4096]
  const float* w2 = (const float*)d_in[4];  // [8, 4096, 2048]
  const float* b2 = (const float*)d_in[5];  // [8, 2048]
  float* out = (float*)d_out;               // [4096, 2048]

  // ws layout: combine [4096*8] f32, then H scratch [4096*4096] f32 (~67 MB)
  float* combine = (float*)d_ws;
  float* H = combine + (size_t)TOKENS * NEXP;

  router_kernel<<<TOKENS / 256, 256, 0, stream>>>(x, rw, combine);

  int outN = TOKENS * D_MODEL;
  zero_kernel<<<(outN + 255) / 256, 256, 0, stream>>>(out, outN);

  for (int e = 0; e < NEXP; ++e) {
    const float* w1e = w1 + (size_t)e * D_MODEL * D_HID;
    const float* b1e = b1 + (size_t)e * D_HID;
    const float* w2e = w2 + (size_t)e * D_HID * D_MODEL;
    const float* b2e = b2 + (size_t)e * D_MODEL;

    // H = gelu(x @ w1[e] + b1[e]) : M=4096, N=4096, K=2048
    dim3 g1(TOKENS / 128, D_HID / 128);
    moe_gemm_kernel<0><<<g1, 256, 0, stream>>>(x, w1e, b1e, nullptr, e, H,
                                               TOKENS, D_HID, D_MODEL);

    // out += combine[:,e] * (H @ w2[e] + b2[e]) : M=4096, N=2048, K=4096
    dim3 g2(TOKENS / 128, D_MODEL / 128);
    moe_gemm_kernel<1><<<g2, 256, 0, stream>>>(H, w2e, b2e, combine, e, out,
                                               TOKENS, D_MODEL, D_HID);
  }
}
